// ShiftedWindowMSA_19516331393725
// MI455X (gfx1250) — matrix operand order repaired and checked by reading, not compiled, not run on hardware
//
#include <hip/hip_runtime.h>
#include <hip/hip_bf16.h>

// ---------------------------------------------------------------------------
// Swin Shifted-Window MSA for gfx1250 (MI455X), wave32.
// WMMA: v_wmma_f32_16x16x32_f16 for all three matmul phases.
// Async: global_load_async_to_lds_b128 (+ s_wait_asynccnt) for pure-f16
//        staging copies (attention q/k tiles, projection A tile).
// ---------------------------------------------------------------------------

typedef __attribute__((ext_vector_type(16))) _Float16 v16h;
typedef __attribute__((ext_vector_type(8)))  float    v8f;

#define WMMA_F16(a, b, c) \
    __builtin_amdgcn_wmma_f32_16x16x32_f16(false, (a), false, (b), (short)0, (c), false, false)

// Async copy of 16 bytes global -> LDS (per active lane). ldsOff is the byte
// offset within the wave's LDS allocation (low 32 bits of a generic pointer
// to a __shared__ object, per the CDNA5 flat-address aperture mapping).
__device__ inline void async_copy_b128(unsigned ldsOff, const void* gaddr) {
    asm volatile("global_load_async_to_lds_b128 %0, %1, off"
                 :: "v"(ldsOff), "v"((unsigned long long)(uintptr_t)gaddr)
                 : "memory");
}
__device__ inline void wait_asynccnt0() {
    asm volatile("s_wait_asynccnt 0" ::: "memory");
}
__device__ inline unsigned lds_off(const void* p) {
    return (unsigned)(uintptr_t)p;   // low 32 bits = LDS offset
}

// A-matrix fragment (16x32 f16, M x K). Per ISA layout:
// element j of lane L holds K = kbase + (j/8)*16 + ((L>=16)?8:0) + (j%8), M = L%16.
__device__ inline v16h load_frag_a(const _Float16* row, int lane, int kbase) {
    union { v16h v; uint4 q[2]; } u;
    int klo = kbase + ((lane & 16) ? 8 : 0);
    u.q[0] = *(const uint4*)(row + klo);
    u.q[1] = *(const uint4*)(row + klo + 16);
    return u.v;
}

// B-matrix fragment (32x16 f16, K x N), staged in LDS transposed as [N][K]:
// element j of lane L holds K = kbase + (j/8)*16 + ((L>=16)?8:0) + (j%8), N = L%16.
__device__ inline v16h load_frag_b(const _Float16* rowN, int lane, int kbase) {
    union { v16h v; uint4 q[2]; } u;
    int kb = kbase + ((lane & 16) ? 8 : 0);
    u.q[0] = *(const uint4*)(rowN + kb);
    u.q[1] = *(const uint4*)(rowN + kb + 16);
    return u.v;
}

// ---------------------------------------------------------------------------
// Kernel 1: qkv = x @ Wqkv^T + b, split into Q/K/V (feature f -> tensor f%3,
// channel f/3), stored f16. Tiles: M=128, N=64, 8 waves (4x2), K step 32.
// (f32 -> f16 conversion in flight, so VGPR staging, not async.)
// ---------------------------------------------------------------------------
__global__ __launch_bounds__(256) void qkv_gemm_kernel(
    const float* __restrict__ x, const float* __restrict__ Wq,
    const float* __restrict__ bq,
    _Float16* __restrict__ Q, _Float16* __restrict__ K, _Float16* __restrict__ V)
{
    __shared__ _Float16 Al[128][32];
    __shared__ _Float16 Bl[64][32];

    const int tid   = threadIdx.x;
    const int lane  = tid & 31;
    const int wid   = tid >> 5;
    const int waveM = wid >> 1;      // 0..3 -> 32-row strip
    const int waveN = wid & 1;       // 0..1 -> 32-col strip
    const long Mbase = (long)blockIdx.x * 128;
    const int  Nbase = blockIdx.y * 64;

    v8f acc[2][2] = {};

    const int kq = tid & 7;          // float4 within K-chunk
    const int rb = tid >> 3;         // 0..31

    for (int k0 = 0; k0 < 192; k0 += 32) {
        // stage A (x tile, f32 -> f16)
        for (int p = 0; p < 4; ++p) {
            int row = p * 32 + rb;
            float4 xv = *(const float4*)(x + (Mbase + row) * 192 + k0 + kq * 4);
            _Float16* d = &Al[row][kq * 4];
            d[0] = (_Float16)xv.x; d[1] = (_Float16)xv.y;
            d[2] = (_Float16)xv.z; d[3] = (_Float16)xv.w;
        }
        // stage B (Wqkv rows, f32 -> f16), BT layout [N][K]
        for (int p = 0; p < 2; ++p) {
            int n = p * 32 + rb;
            float4 wv = *(const float4*)(Wq + (long)(Nbase + n) * 192 + k0 + kq * 4);
            _Float16* d = &Bl[n][kq * 4];
            d[0] = (_Float16)wv.x; d[1] = (_Float16)wv.y;
            d[2] = (_Float16)wv.z; d[3] = (_Float16)wv.w;
        }
        __syncthreads();

        v16h af[2], bf[2];
        for (int sm = 0; sm < 2; ++sm)
            af[sm] = load_frag_a(&Al[waveM * 32 + sm * 16 + (lane & 15)][0], lane, 0);
        for (int sn = 0; sn < 2; ++sn)
            bf[sn] = load_frag_b(&Bl[waveN * 32 + sn * 16 + (lane & 15)][0], lane, 0);
        for (int sm = 0; sm < 2; ++sm)
            for (int sn = 0; sn < 2; ++sn)
                acc[sm][sn] = WMMA_F16(af[sm], bf[sn], acc[sm][sn]);
        __syncthreads();
    }

    // epilogue: add bias, de-interleave into Q/K/V, store f16
    for (int sm = 0; sm < 2; ++sm) {
        for (int sn = 0; sn < 2; ++sn) {
            int f = Nbase + waveN * 32 + sn * 16 + (lane & 15);
            float bias = bq[f];
            int sel = f % 3;
            int c   = f / 3;
            _Float16* dst = (sel == 0) ? Q : ((sel == 1) ? K : V);
            long mb = Mbase + waveM * 32 + sm * 16 + ((lane >> 4) << 3);
            for (int r = 0; r < 8; ++r)
                dst[(mb + r) * 192 + c] = (_Float16)(acc[sm][sn][r] + bias);
        }
    }
}

// ---------------------------------------------------------------------------
// Kernel 2: windowed attention. One block (128 thr = 4 waves) per
// (batch, head, wy, wx). 49 tokens padded to 64, head_dim = 32.
// q/k tiles staged with async global->LDS copies (pure f16 copies).
// ---------------------------------------------------------------------------
__global__ __launch_bounds__(128) void attn_kernel(
    const _Float16* __restrict__ Q, const _Float16* __restrict__ Kt,
    const _Float16* __restrict__ Vt, const float* __restrict__ Brel,
    _Float16* __restrict__ P)
{
    __shared__ _Float16 ql[64][32];     // q tile, row-major [token][dim]
    __shared__ _Float16 kl[64][32];     // k tile (== B^T for scores GEMM)
    __shared__ _Float16 vTl[32][64];    // v^T  (== B^T for PV GEMM)
    __shared__ _Float16 pl[64][64];     // softmax probabilities (A of PV GEMM)
    __shared__ float    sc[64][64];     // raw / staged scores
    __shared__ float    brel[169];      // 13x13 relative bias table

    const int bid = blockIdx.x;
    const int wx = bid & 7;
    const int wy = (bid >> 3) & 7;
    const int rest = bid >> 6;          // 0..191
    const int h = rest % 6;
    const int b = rest / 6;

    const int tid = threadIdx.x, lane = tid & 31, w = tid >> 5;

    for (int i = tid; i < 169; i += 128) brel[i] = Brel[i];

    // zero the 15 padding rows (49..63) of ql/kl with plain DS stores
    for (int idx = tid; idx < 15 * 32; idx += 128) {
        int row = 49 + (idx >> 5), col = idx & 31;
        ql[row][col] = (_Float16)0.f;
        kl[row][col] = (_Float16)0.f;
    }

    // stage q, k via async copies; v transposed via VGPRs; roll shift -4
    const int part = tid & 3;           // 8-half (16B) chunk
    const int rb   = tid >> 2;          // 0..31
    for (int p = 0; p < 2; ++p) {
        int row = p * 32 + rb;
        uint4 vv = {0, 0, 0, 0};
        if (row < 49) {
            int m1 = row / 7, m2 = row % 7;
            int y  = (wy * 7 + m1 + 4) % 56;
            int xx = (wx * 7 + m2 + 4) % 56;
            long base = ((long)(b * 3136 + y * 56 + xx)) * 192 + h * 32 + part * 8;
            async_copy_b128(lds_off(&ql[row][part * 8]), Q  + base);
            async_copy_b128(lds_off(&kl[row][part * 8]), Kt + base);
            vv = *(const uint4*)(Vt + base);
        }
        const _Float16* vh = (const _Float16*)&vv;
        for (int j = 0; j < 8; ++j) vTl[part * 8 + j][row] = vh[j];
    }
    wait_asynccnt0();
    __syncthreads();

    // scores = q @ k^T : wave w computes rows [16w,16w+16) x all 64 cols
    {
        v16h a = load_frag_a(&ql[w * 16 + (lane & 15)][0], lane, 0);
        for (int t = 0; t < 4; ++t) {
            v16h bf = load_frag_b(&kl[t * 16 + (lane & 15)][0], lane, 0);
            v8f c = {};
            c = WMMA_F16(a, bf, c);
            int n  = t * 16 + (lane & 15);
            int mb = w * 16 + ((lane >> 4) << 3);
            for (int r = 0; r < 8; ++r) sc[mb + r][n] = c[r];
        }
    }
    __syncthreads();

    // softmax with scale + relative bias + shift masks (one row per thread)
    if (tid < 64) {
        int i = tid;
        if (i >= 49) {
            for (int j = 0; j < 64; ++j) pl[i][j] = (_Float16)0.f;
        } else {
            const float invs = 0.17677669529663687f;  // 1/sqrt(32)
            const bool lastrow = (wy == 7), lastcol = (wx == 7);
            const int im7 = i % 7;
            float mx = -3.0e38f;
            for (int j = 0; j < 49; ++j) {
                int xi = (j - i) / 7;                 // trunc toward zero
                int yi = (j % 7) - im7;
                if (xi < 0) xi += 13;
                if (yi < 0) yi += 13;
                float s = sc[i][j] * invs + brel[xi * 13 + yi];
                if (lastrow && ((i >= 28) != (j >= 28)))        s -= 1.0e30f;
                if (lastcol && ((im7 >= 4) != ((j % 7) >= 4)))  s -= 1.0e30f;
                sc[i][j] = s;
                mx = fmaxf(mx, s);
            }
            float sum = 0.f;
            for (int j = 0; j < 49; ++j) {
                float e = __expf(sc[i][j] - mx);
                sum += e;
                sc[i][j] = e;
            }
            float rs = 1.f / sum;
            for (int j = 0; j < 49; ++j)  pl[i][j] = (_Float16)(sc[i][j] * rs);
            for (int j = 49; j < 64; ++j) pl[i][j] = (_Float16)0.f;
        }
    }
    __syncthreads();

    // out = p @ v : wave w rows x 32 dims; reverse roll +3 on store
    {
        v16h a0 = load_frag_a(&pl[w * 16 + (lane & 15)][0], lane, 0);
        v16h a1 = load_frag_a(&pl[w * 16 + (lane & 15)][0], lane, 32);
        for (int nt = 0; nt < 2; ++nt) {
            const _Float16* vrow = &vTl[nt * 16 + (lane & 15)][0];
            v8f c = {};
            c = WMMA_F16(a0, load_frag_b(vrow, lane, 0),  c);
            c = WMMA_F16(a1, load_frag_b(vrow, lane, 32), c);
            int dim = nt * 16 + (lane & 15);
            int mb  = w * 16 + ((lane >> 4) << 3);
            for (int r = 0; r < 8; ++r) {
                int m = mb + r;
                if (m < 49) {
                    int m1 = m / 7, m2 = m % 7;
                    int y  = (wy * 7 + m1 + 3) % 56;
                    int xx = (wx * 7 + m2 + 3) % 56;
                    P[((long)(b * 3136 + y * 56 + xx)) * 192 + h * 32 + dim] =
                        (_Float16)c[r];
                }
            }
        }
    }
}

// ---------------------------------------------------------------------------
// Kernel 3: out = P @ Wout^T + b_out (f16 A, f32 output). M=128, N=64 tiles.
// A tile (already f16) staged with async global->LDS copies.
// ---------------------------------------------------------------------------
__global__ __launch_bounds__(256) void proj_gemm_kernel(
    const _Float16* __restrict__ P, const float* __restrict__ Wo,
    const float* __restrict__ bo, float* __restrict__ out)
{
    __shared__ _Float16 Al[128][32];
    __shared__ _Float16 Bl[64][32];

    const int tid   = threadIdx.x;
    const int lane  = tid & 31;
    const int wid   = tid >> 5;
    const int waveM = wid >> 1;
    const int waveN = wid & 1;
    const long Mbase = (long)blockIdx.x * 128;
    const int  Nbase = blockIdx.y * 64;

    v8f acc[2][2] = {};

    for (int k0 = 0; k0 < 192; k0 += 32) {
        // stage A: f16 rows, async 16B copies straight into LDS
        {
            int part = tid & 3, r = tid >> 2;       // r 0..63
            for (int p = 0; p < 2; ++p) {
                int row = p * 64 + r;
                async_copy_b128(lds_off(&Al[row][part * 8]),
                                P + (Mbase + row) * 192 + k0 + part * 8);
            }
        }
        // stage B: Wout rows f32 -> f16 (conversion => VGPR staging)
        {
            int kq = tid & 7, r = tid >> 3;         // r 0..31
            for (int p = 0; p < 2; ++p) {
                int n = p * 32 + r;
                float4 wv = *(const float4*)(Wo + (long)(Nbase + n) * 192 + k0 + kq * 4);
                _Float16* d = &Bl[n][kq * 4];
                d[0] = (_Float16)wv.x; d[1] = (_Float16)wv.y;
                d[2] = (_Float16)wv.z; d[3] = (_Float16)wv.w;
            }
        }
        wait_asynccnt0();
        __syncthreads();

        v16h af[2], bf[2];
        for (int sm = 0; sm < 2; ++sm)
            af[sm] = load_frag_a(&Al[waveM * 32 + sm * 16 + (lane & 15)][0], lane, 0);
        for (int sn = 0; sn < 2; ++sn)
            bf[sn] = load_frag_b(&Bl[waveN * 32 + sn * 16 + (lane & 15)][0], lane, 0);
        for (int sm = 0; sm < 2; ++sm)
            for (int sn = 0; sn < 2; ++sn)
                acc[sm][sn] = WMMA_F16(af[sm], bf[sn], acc[sm][sn]);
        __syncthreads();
    }

    for (int sm = 0; sm < 2; ++sm) {
        for (int sn = 0; sn < 2; ++sn) {
            int f = Nbase + waveN * 32 + sn * 16 + (lane & 15);
            float bias = bo[f];
            long mb = Mbase + waveM * 32 + sm * 16 + ((lane >> 4) << 3);
            for (int r = 0; r < 8; ++r)
                out[(mb + r) * 192 + f] = acc[sm][sn][r] + bias;
        }
    }
}

// ---------------------------------------------------------------------------
extern "C" void kernel_launch(void* const* d_in, const int* in_sizes, int n_in,
                              void* d_out, int out_size, void* d_ws, size_t ws_size,
                              hipStream_t stream) {
    const float* x    = (const float*)d_in[0];   // (32, 3136, 192)
    const float* Wqkv = (const float*)d_in[1];   // (576, 192)
    const float* bqkv = (const float*)d_in[2];   // (576,)
    const float* Wout = (const float*)d_in[3];   // (192, 192)
    const float* bout = (const float*)d_in[4];   // (192,)
    const float* Brel = (const float*)d_in[5];   // (13, 13)
    float* out = (float*)d_out;

    const size_t elems = (size_t)32 * 3136 * 192;     // per-tensor f16 elements
    _Float16* Q = (_Float16*)d_ws;
    _Float16* K = Q + elems;
    _Float16* V = K + elems;
    _Float16* P = V + elems;                          // total 4*elems*2 = ~147 MB

    // Phase 1: QKV projection (M=100352 -> 784 M-tiles, 576 -> 9 N-tiles)
    qkv_gemm_kernel<<<dim3(784, 9), 256, 0, stream>>>(x, Wqkv, bqkv, Q, K, V);

    // Phase 2: attention over 32 batches * 6 heads * 64 windows
    attn_kernel<<<dim3(12288), 128, 0, stream>>>(Q, K, V, Brel, P);

    // Phase 3: output projection (192 -> 3 N-tiles)
    proj_gemm_kernel<<<dim3(784, 3), 256, 0, stream>>>(P, Wout, bout, out);
}
